// DPBlockVFAStandard_81647328297391
// MI455X (gfx1250) — matrix operand order repaired and checked by reading, not compiled, not run on hardware
//
#include <hip/hip_runtime.h>
#include <hip/hip_bf16.h>

typedef __attribute__((ext_vector_type(16))) _Float16     v16h;
typedef __attribute__((ext_vector_type(8)))  float        v8f;
typedef __attribute__((ext_vector_type(4)))  unsigned int v4u;
typedef __attribute__((ext_vector_type(8)))  unsigned int v8u;

#define EMBED  16
#define CIN    32
#define NOFF   27
#define KTOT   (CIN * NOFF)      // 864
#define VOL64  (64 * 64 * 64)
#define VOL66  (66 * 66 * 66)

#define WAVES_PER_BLOCK 4
#define LDS_PER_WAVE    (9 * 18 * 32)   // [tap9][iz 0..17][cin 0..31] f16

__device__ __forceinline__ int clampi(int v, int lo, int hi) {
    return v < lo ? lo : (v > hi ? hi : v);
}

// ---------------------------------------------------------------------------
// Weight prep: conv_w [16][32][27] f32 -> W16 [16][off*32 + cin] f16
// Offset-major K; cin stored in contiguous K order (k = 16*khalf + s), same as B, so
// the conv kernel's A load is one contiguous 32B (v16h) global load per tap.
// ---------------------------------------------------------------------------
__global__ void prep_weights_kernel(const float* __restrict__ w,
                                    _Float16* __restrict__ w16) {
    int idx = blockIdx.x * blockDim.x + threadIdx.x;
    if (idx >= EMBED * KTOT) return;
    int r     = idx / KTOT;
    int t     = idx % KTOT;
    int off   = t >> 5;
    int u     = t & 31;
    int khalf = u >> 4;
    int h     = u & 15;
    int cin   = h + 16 * khalf;   // contiguous K order (k = 16*khalf + s), matches B
    w16[idx] = (_Float16)w[(r * CIN + cin) * NOFF + off];
}

// ---------------------------------------------------------------------------
// Conv3d 32->16 as WMMA GEMM with LDS halo staging (one wave = 16-voxel strip,
// 27 accumulating v_wmma_f32_16x16x32_f16).
// MODE   0: zero pad on 64^3 (fixed)   1: 66^3 grid, edge+zero pad (moving)
// LAYOUT 0: channel-major [16][OUTD^3] 1: voxel-major [OUTD^3][16] (packed b32)
// ---------------------------------------------------------------------------
template <int MODE, int OUTD, int NDT, int LAYOUT>
__global__ void conv_wmma_kernel(const float* __restrict__ src,   // [32][64^3]
                                 const float* __restrict__ bias,  // [16]
                                 const _Float16* __restrict__ w16,// [16][864]
                                 _Float16* __restrict__ dst,
                                 int nTiles) {
    __shared__ __align__(64) _Float16 lds[WAVES_PER_BLOCK][LDS_PER_WAVE];

    const int lane  = threadIdx.x & 31;
    const int wave  = threadIdx.x >> 5;
    const int tile  = blockIdx.x * WAVES_PER_BLOCK + wave;
    if (tile >= nTiles) return;                 // whole-wave uniform exit

    const int col   = lane & 15;                // B/D column (voxel in strip)
    const int khalf = lane >> 4;                // K half selector

    const int td = tile % NDT;
    const int x  = (tile / NDT) % OUTD;
    const int y  =  tile / (NDT * OUTD);
    const int d0 = td * 16;

    _Float16* myLds = &lds[wave][0];

    // ---- Stage halo block: 288 rows = (cin) x (tap9), 18 z each; rolled
    // loops, branch-free bounds (clamped in-range load + select-zero).
    #pragma unroll 1
    for (int it = 0; it < 9; ++it) {
        const int r  = lane + 32 * it;          // 288 = 32*9 exact
        const int c  = r / 9;
        const int t9 = r % 9;
        const int ii = t9 / 3;
        const int jj = t9 % 3;
        const int gy = y + ii - 1;
        const int gx = x + jj - 1;

        bool rowOK;
        int  sy, sx;
        if (MODE == 0) {
            rowOK = (unsigned)gy < 64u && (unsigned)gx < 64u;
            sy = clampi(gy, 0, 63);
            sx = clampi(gx, 0, 63);
        } else {
            rowOK = (unsigned)gy < 66u && (unsigned)gx < 66u;
            sy = clampi(gy - 1, 0, 63);
            sx = clampi(gx - 1, 0, 63);
        }
        const float* p = src + (size_t)c * VOL64 + ((size_t)(sy * 64 + sx) * 64);
        _Float16* q = myLds + (size_t)t9 * (18 * 32) + c;

        #pragma unroll 1
        for (int iz = 0; iz < 18; ++iz) {
            const int gz = d0 + iz - 1;
            bool zOK;
            int  sz;
            if (MODE == 0) {
                zOK = (unsigned)gz < 64u;
                sz  = clampi(gz, 0, 63);
            } else {
                zOK = (unsigned)gz < 66u;
                sz  = clampi(gz - 1, 0, 63);
            }
            float v   = p[sz];
            float val = (rowOK && zOK) ? v : 0.0f;
            q[(size_t)iz * 32] = (_Float16)val;
        }
    }
    // Private LDS slice per wave; same-wave DS ordering, no barrier needed.

    // ---- 27 accumulating WMMAs (A: contiguous global b128, B: 32B LDS read)
    const _Float16* wrow = w16 + (size_t)col * KTOT + khalf * 16;

    v8f acc = {};
    #pragma unroll
    for (int off = 0; off < NOFF; ++off) {
        const int t9 = off / 3;
        const int k  = off % 3;
        v16h a = *(const v16h*)(wrow + off * 32);
        v16h b = *(const v16h*)(myLds + ((size_t)t9 * 18 + (col + k)) * 32
                                      + khalf * 16);
        acc = __builtin_amdgcn_wmma_f32_16x16x32_f16(
            false, a, false, b, (short)0, acc, false, false);
    }

    const int d = d0 + col;
    if (d < OUTD) {
        const size_t vox = ((size_t)y * OUTD + x) * OUTD + d;
        if (LAYOUT == 0) {
            const size_t vol = (size_t)OUTD * OUTD * OUTD;
            #pragma unroll
            for (int g = 0; g < 8; ++g) {       // D layout: row = g + 8*khalf
                int row = g + 8 * khalf;
                dst[(size_t)row * vol + vox] = (_Float16)(acc[g] + bias[row]);
            }
        } else {
            // voxel-major [vox][16]: rows 8*khalf..8*khalf+7 are 8 consecutive
            // halves -> pack into 4 dword stores.
            unsigned* dp = (unsigned*)dst + vox * 8 + khalf * 4;
            #pragma unroll
            for (int tq = 0; tq < 4; ++tq) {
                int row0 = 2 * tq + 8 * khalf;
                _Float16 h0 = (_Float16)(acc[2 * tq]     + bias[row0]);
                _Float16 h1 = (_Float16)(acc[2 * tq + 1] + bias[row0 + 1]);
                unsigned u = (unsigned)__builtin_bit_cast(unsigned short, h0)
                           | ((unsigned)__builtin_bit_cast(unsigned short, h1) << 16);
                dp[tq] = u;
            }
        }
    }
}

// ---------------------------------------------------------------------------
// Attention with TDM-staged moving embeddings.
// Block = 256 threads = one 4(y) x 4(x) x 16(z) voxel brick.  The brick's
// mov_emb halo tile ([6][6][18][16e] f16, all in-range on the 66^3 grid) is
// DMA'd into LDS by the Tensor Data Mover as one 3D tile:
//   dim0 = 18z*16e = 288 contiguous elements, dim1 = 6 (x, stride 1056),
//   dim2 = 6 (y, stride 69696); data_size = 2B.
// Wave 0 issues tensor_load_to_lds (D# in SGPRs), waits TENSORcnt, block
// barriers, then each thread does 27 taps of 32B-aligned v16h LDS reads.
// ---------------------------------------------------------------------------
__global__ void attn_disp_kernel(const _Float16* __restrict__ fixed_emb, // [16][64^3]
                                 const _Float16* __restrict__ mov_emb,   // [66^3][16]
                                 float* __restrict__ out) {              // [3][64^3]
    __shared__ __align__(64) _Float16 tile[6 * 6 * 288];   // 20736 B

    const int b  = blockIdx.x;
    const int bz = b & 3;             // 4 z-bricks of 16
    const int bx = (b >> 2) & 15;     // 16 x-bricks of 4
    const int by = b >> 6;            // 16 y-bricks of 4
    const int z0 = bz * 16, x0 = bx * 4, y0 = by * 4;

    // ---- Tensor DMA descriptor (uniform -> SGPRs via "s" constraints)
    const unsigned ldsAddr = (unsigned)(uintptr_t)&tile[0];
    const unsigned long long gaddr =
        (unsigned long long)(const void*)(mov_emb +
            ((size_t)(y0 * 66 + x0) * 66 + z0) * 16);

    v4u g0;
    g0[0] = 1u;                                   // count=1, user descriptor
    g0[1] = ldsAddr;                              // LDS byte address
    g0[2] = (unsigned)gaddr;                      // global_addr[31:0]
    g0[3] = (unsigned)((gaddr >> 32) & 0x1FFFFFFu) | (2u << 30);  // [56:32]|type=2

    v8u g1;
    g1[0] = 1u << 16;                             // data_size=1 (2 bytes)
    g1[1] = (1056u & 0xFFFFu) << 16;              // tensor_dim0 lo16 (66*16)
    g1[2] = (1056u >> 16) | (66u << 16);          // dim0 hi16 | tensor_dim1 lo16
    g1[3] = (288u << 16);                         // dim1 hi16 | tile_dim0 = 288
    g1[4] = 6u | (6u << 16);                      // tile_dim1 = 6, tile_dim2 = 6
    g1[5] = 1056u;                                // tensor_dim0_stride lo32
    g1[6] = ((69696u & 0xFFFFu) << 16);           // stride0 hi16 | stride1 lo16
    g1[7] = 69696u >> 16;                         // tensor_dim1_stride [47:16]

    v4u g2;
    g2[0] = 66u;                                  // tensor_dim2
    g2[1] = 0u; g2[2] = 0u; g2[3] = 0u;           // no dim3
    v4u g3 = (v4u){0u, 0u, 0u, 0u};

    if (threadIdx.x < 32) {                       // wave 0 issues the DMA
        asm volatile("tensor_load_to_lds %0, %1, %2, %3"
                     :: "s"(g0), "s"(g1), "s"(g2), "s"(g3)
                     : "memory");
        __builtin_amdgcn_s_wait_tensorcnt(0);
    }
    __syncthreads();

    // ---- Per-voxel attention from LDS
    const int t  = threadIdx.x;
    const int lz = t & 15, lx = (t >> 4) & 3, ly = t >> 6;
    const int z = z0 + lz, x = x0 + lx, y = y0 + ly;
    const int v = (y * 64 + x) * 64 + z;

    float q[EMBED];
    #pragma unroll
    for (int e = 0; e < EMBED; ++e)
        q[e] = (float)fixed_emb[(size_t)e * VOL64 + v];

    float sc[NOFF];
    float mx = -1e30f;
    #pragma unroll
    for (int p = 0; p < NOFF; ++p) {
        const int i = p / 9, j = (p / 3) % 3, k = p % 3;
        const v16h m = *(const v16h*)&tile[(((ly + i) * 6 + (lx + j)) * 18
                                            + (lz + k)) * 16];
        float s = 0.0f;
        #pragma unroll
        for (int e = 0; e < EMBED; ++e)
            s = __builtin_fmaf(q[e], (float)m[e], s);
        s *= 0.25f;                     // / TEMP (= 4.0)
        sc[p] = s;
        mx = fmaxf(mx, s);
    }

    float sum = 0.0f, dh = 0.0f, dw = 0.0f, dd = 0.0f;
    #pragma unroll
    for (int p = 0; p < NOFF; ++p) {
        float e = __expf(sc[p] - mx);
        sum += e;
        dh += e * (float)(p / 9 - 1);
        dw += e * (float)((p / 3) % 3 - 1);
        dd += e * (float)(p % 3 - 1);
    }
    float inv = 1.0f / sum;
    out[v]             = dh * inv;     // H component
    out[VOL64 + v]     = dw * inv;     // W component
    out[2 * VOL64 + v] = dd * inv;     // D component
}

// ---------------------------------------------------------------------------
extern "C" void kernel_launch(void* const* d_in, const int* in_sizes, int n_in,
                              void* d_out, int out_size, void* d_ws, size_t ws_size,
                              hipStream_t stream) {
    const float* feat_moving = (const float*)d_in[0];
    const float* feat_fixed  = (const float*)d_in[1];
    const float* conv_w      = (const float*)d_in[2];
    const float* conv_b      = (const float*)d_in[3];
    float* out = (float*)d_out;

    // Workspace:
    //   W16       : 16*864*2  (reserve 32 KiB)
    //   fixed_emb : [16][64^3] f16 = 8,388,608 B
    //   mov_emb   : [66^3][16] f16 = 9,199,872 B   -> ~17.6 MB total
    char* ws = (char*)d_ws;
    _Float16* w16       = (_Float16*)ws;
    _Float16* fixed_emb = (_Float16*)(ws + 32768);
    _Float16* mov_emb   = (_Float16*)(ws + 32768 + (size_t)EMBED * VOL64 * 2);

    prep_weights_kernel<<<(EMBED * KTOT + 255) / 256, 256, 0, stream>>>(conv_w, w16);

    // Fixed path: 64^3, 4 d-tiles, 16384 strips, channel-major output
    conv_wmma_kernel<0, 64, 4, 0>
        <<<16384 / WAVES_PER_BLOCK, 32 * WAVES_PER_BLOCK, 0, stream>>>(
            feat_fixed, conv_b, w16, fixed_emb, 16384);

    // Moving path: 66^3, 5 d-tiles, 21780 strips, voxel-major output for TDM
    const int ntm = 66 * 66 * 5;
    conv_wmma_kernel<1, 66, 5, 1>
        <<<(ntm + WAVES_PER_BLOCK - 1) / WAVES_PER_BLOCK,
           32 * WAVES_PER_BLOCK, 0, stream>>>(
            feat_moving, conv_b, w16, mov_emb, ntm);

    // Attention: 1024 bricks of 4x4x16 voxels, TDM-staged LDS tiles
    attn_disp_kernel<<<1024, 256, 0, stream>>>(fixed_emb, mov_emb, out);
}
